// SCNN_LaneDetection_45303315038734
// MI455X (gfx1250) — matrix operand order repaired and checked by reading, not compiled, not run on hardware
//
#include <hip/hip_runtime.h>
#include <hip/hip_bf16.h>

typedef _Float16 half_t;
typedef __attribute__((ext_vector_type(16))) _Float16 v16h;
typedef __attribute__((ext_vector_type(8)))  _Float16 v8h;
typedef __attribute__((ext_vector_type(8)))  float    v8f;
typedef __attribute__((ext_vector_type(4)))  int      v4i;

#define CTW 40    // generic kernel LDS K-stride (32 + 8 pad)
#define CTW2 72   // fast kernel LDS K-stride (64 + 8 pad; 144B keeps 16B alignment)

#if defined(__gfx1250__) && __has_builtin(__builtin_amdgcn_global_load_async_to_lds_b128) && \
    __has_builtin(__builtin_amdgcn_s_wait_asynccnt)
#define HAVE_ASYNC_LDS 1
#warning CDNA5: async-to-LDS double-buffered pipeline ENABLED
#else
#define HAVE_ASYNC_LDS 0
#endif

__device__ __forceinline__ v16h frag16(const half_t* p, int o0, int o1) {
  v8h lo = *(const v8h*)(p + o0);
  v8h hi = *(const v8h*)(p + o1);
  return __builtin_shufflevector(lo, hi, 0,1,2,3,4,5,6,7,8,9,10,11,12,13,14,15);
}

__device__ __forceinline__ v16h zero16h() {
  v16h z = {(half_t)0,(half_t)0,(half_t)0,(half_t)0,(half_t)0,(half_t)0,(half_t)0,(half_t)0,
            (half_t)0,(half_t)0,(half_t)0,(half_t)0,(half_t)0,(half_t)0,(half_t)0,(half_t)0};
  return z;
}

#if HAVE_ASYNC_LDS
__device__ __forceinline__ void async_cp16(const half_t* g, half_t* l) {
  __builtin_amdgcn_global_load_async_to_lds_b128(
      (__attribute__((address_space(1))) v4i*)g,
      (__attribute__((address_space(3))) v4i*)l, 0, 0);
}
#else
__device__ __forceinline__ void async_cp16(const half_t* g, half_t* l) {
  *(v8h*)l = *(const v8h*)g;
}
#endif

__device__ __forceinline__ int clampi(int v, int lo, int hi) {
  return v < lo ? lo : (v > hi ? hi : v);
}

// ---------------- packing / utility kernels ----------------

__global__ void pack_input_kernel(const float* __restrict__ src, half_t* __restrict__ dst,
                                  int N, int H, int W, int C) {
  size_t idx = (size_t)blockIdx.x * blockDim.x + threadIdx.x;
  size_t total = (size_t)N * H * W * C;
  if (idx >= total) return;
  int c = idx % C; size_t r = idx / C;
  int x = r % W; r /= W;
  int y = r % H; int n = (int)(r / H);
  dst[idx] = (half_t)src[(((size_t)n * C + c) * H + y) * W + x];
}

// OIHW f32 -> [Cout][K] f16, K = (kh*ks+kw)*Cin + ci
__global__ void pack_convw_kernel(const float* __restrict__ src, half_t* __restrict__ dst,
                                  int Cout, int Cin, int ks) {
  int Ktot = Cin * ks * ks;
  size_t idx = (size_t)blockIdx.x * blockDim.x + threadIdx.x;
  size_t total = (size_t)Cout * Ktot;
  if (idx >= total) return;
  int oc = (int)(idx / Ktot);
  int k  = (int)(idx % Ktot);
  int rr = k / Cin, ci = k - rr * Cin;
  int kh = rr / ks, kw = rr - kh * ks;
  dst[idx] = (half_t)src[(((size_t)oc * Cin + ci) * ks + kh) * ks + kw];
}

__global__ void pack_mpw_kernel(const float* __restrict__ src, half_t* __restrict__ dst) {
  int idx = blockIdx.x * blockDim.x + threadIdx.x;
  if (idx >= 128 * 1152) return;
  int oc = idx / 1152;
  int k  = idx % 1152;
  int tap = k >> 7, ci = k & 127;
  dst[idx] = (half_t)src[((size_t)oc * 128 + ci) * 9 + tap];
}

__global__ void maxpool_kernel(const half_t* __restrict__ in, half_t* __restrict__ out,
                               int N, int Hin, int Win, int C) {
  int Ho = Hin >> 1, Wo = Win >> 1;
  size_t idx = (size_t)blockIdx.x * blockDim.x + threadIdx.x;
  size_t total = (size_t)N * Ho * Wo * C;
  if (idx >= total) return;
  int c = idx % C; size_t r = idx / C;
  int x = r % Wo; r /= Wo;
  int y = r % Ho; int n = (int)(r / Ho);
  const half_t* p = in + (((size_t)n * Hin + 2 * y) * Win + 2 * x) * C + c;
  float m0 = (float)p[0];
  float m1 = (float)p[(size_t)C];
  float m2 = (float)p[(size_t)Win * C];
  float m3 = (float)p[(size_t)Win * C + C];
  out[idx] = (half_t)fmaxf(fmaxf(m0, m1), fmaxf(m2, m3));
}

// ---------------- generic implicit-GEMM conv (any Cin) — used only for conv0 ----------------
__global__ __launch_bounds__(256)
void conv_wmma_kernel(const half_t* __restrict__ in, const half_t* __restrict__ wpk,
                      const float* __restrict__ gp, const float* __restrict__ bp,
                      const float* __restrict__ mp, const float* __restrict__ vp,
                      half_t* __restrict__ out,
                      int H, int W, int Cin, int Cout, int ks, int dil, int relu) {
  __shared__ __align__(16) half_t sA[64 * CTW];
  __shared__ __align__(16) half_t sB[32 * CTW];

  const int tid  = threadIdx.x;
  const int lane = tid & 31;
  const int wave = tid >> 5;
  const int pxSub = (wave & 3) * 16;
  const int ocSub = (wave >> 2) * 16;
  const int n = blockIdx.z;
  const int HW = H * W;
  const int pxBlock = blockIdx.x * 64;
  const int ocBlock = blockIdx.y * 32;
  const int Ktot = Cin * ks * ks;
  const int pad = ((ks - 1) >> 1) * dil;

  const half_t* inImg = in + (size_t)n * HW * Cin;

  const int half_id = lane >> 4;
  const int lrow = lane & 15;
  const int offA0 = half_id * 8;
  const int offA1 = 16 + half_id * 8;
  const int offB0 = half_id * 8;
  const int offB1 = 16 + half_id * 8;

  v8f acc = {0.f, 0.f, 0.f, 0.f, 0.f, 0.f, 0.f, 0.f};

  for (int kb = 0; kb < Ktot; kb += 32) {
    #pragma unroll
    for (int i = 0; i < 8; ++i) {
      int e = i * 256 + tid;
      int pxl = e >> 5;
      int kk = e & 31;
      int k = kb + kk;
      half_t val = (half_t)0.0f;
      int px = pxBlock + pxl;
      if (k < Ktot && px < HW) {
        int rr = k / Cin;
        int ci = k - rr * Cin;
        int kh = rr / ks;
        int kw = rr - kh * ks;
        int y = px / W;
        int x = px - y * W;
        int iy = y + kh * dil - pad;
        int ix = x + kw * dil - pad;
        if (iy >= 0 && iy < H && ix >= 0 && ix < W)
          val = inImg[((size_t)iy * W + ix) * Cin + ci];
      }
      sA[pxl * CTW + kk] = val;
    }
    #pragma unroll
    for (int i = 0; i < 4; ++i) {
      int e = i * 256 + tid;
      int kk = e >> 5;
      int oc = e & 31;
      int k = kb + kk;
      half_t wv = (half_t)0.0f;
      if (k < Ktot && (ocBlock + oc) < Cout)
        wv = wpk[(size_t)(ocBlock + oc) * Ktot + k];
      sB[oc * CTW + kk] = wv;
    }
    __syncthreads();
    v16h af = frag16(sA + (pxSub + lrow) * CTW, offA0, offA1);
    v16h bf = frag16(sB + (ocSub + lrow) * CTW, offB0, offB1);
    acc = __builtin_amdgcn_wmma_f32_16x16x32_f16(false, af, false, bf, (short)0, acc,
                                                 false, false);
    __syncthreads();
  }

  int oc = ocBlock + ocSub + lrow;
  if (oc < Cout) {
    float sc = gp[oc] * rsqrtf(vp[oc] + 1e-5f);
    float sh = bp[oc] - mp[oc] * sc;
    half_t* outImg = out + (size_t)n * HW * Cout;
    #pragma unroll
    for (int r = 0; r < 8; ++r) {
      int px = pxBlock + pxSub + half_id * 8 + r;
      if (px < HW) {
        float v = acc[r] * sc + sh;
        if (relu) v = v > 0.0f ? v : 0.0f;
        outImg[(size_t)px * Cout + oc] = (half_t)v;
      }
    }
  }
}

// ---------------- fast implicit-GEMM conv: Cin%64==0, Cout%64==0 ----------------
// block: 256 thr = 8 waves (4 px groups x 2 oc groups); block tile 128 px x 64 oc.
// Stage = one tap x 64 channels; double-buffered async LDS pipeline; each wave
// computes 2x2 WMMA tiles x 2 k-subchunks = 8 WMMAs per stage.
// Staging is divergence-free (clamped addresses); im2col zero-padding is applied
// as a per-lane fragment select in the compute phase.
__global__ __launch_bounds__(256)
void conv_wmma_fast_kernel(const half_t* __restrict__ in, const half_t* __restrict__ wpk,
                           const float* __restrict__ gp, const float* __restrict__ bp,
                           const float* __restrict__ mp, const float* __restrict__ vp,
                           half_t* __restrict__ out,
                           int H, int W, int Cin, int Cout, int ks, int dil, int relu) {
  __shared__ __align__(16) half_t sA[2 * 128 * CTW2];
  __shared__ __align__(16) half_t sB[2 * 64 * CTW2];

  const int tid  = threadIdx.x;
  const int lane = tid & 31;
  const int wave = tid >> 5;
  const int pxW = (wave & 3) * 32;
  const int ocW = (wave >> 2) * 32;
  const int n = blockIdx.z;
  const int HW = H * W;
  const int pxBlock = blockIdx.x * 128;
  const int ocBlock = blockIdx.y * 64;
  const int Ktot = Cin * ks * ks;
  const int ksq = ks * ks;
  const int pad = ((ks - 1) >> 1) * dil;
  const int nCb = Cin >> 6;          // 64-channel blocks per tap
  const int nSteps = ksq * nCb;

  const half_t* inImg = in + (size_t)n * HW * Cin;

  const int half_id = lane >> 4;
  const int lrow = lane & 15;
  const int offA0 = half_id * 8;
  const int offA1 = 16 + half_id * 8;
  const int offB0 = half_id * 8;
  const int offB1 = 16 + half_id * 8;

  // staging geometry: 4 A pixel-rows + 2 B oc-rows per thread, one 16B chunk each
  const int qoff = (tid & 7) * 8;
  int syv[4], sxv[4];
  #pragma unroll
  for (int i = 0; i < 4; ++i) {
    int px = pxBlock + i * 32 + (tid >> 3);
    px = px < HW ? px : HW - 1;          // clamp (zeroed later via frag select)
    int y = px / W;
    syv[i] = y;
    sxv[i] = px - y * W;
  }

  // compute-side pixel coords for fragment masking
  const int pxa0 = pxBlock + pxW + lrow;
  const int pxa1 = pxa0 + 16;
  const int ina0 = pxa0 < HW;
  const int ina1 = pxa1 < HW;
  const int ya0 = pxa0 / W, xa0 = pxa0 - ya0 * W;
  const int ya1 = pxa1 / W, xa1 = pxa1 - ya1 * W;

  v8f acc00 = {0.f,0.f,0.f,0.f,0.f,0.f,0.f,0.f};
  v8f acc01 = acc00, acc10 = acc00, acc11 = acc00;

  // issue one stage's 6 async copies (4 A + 2 B) into buffer bufI
  auto issueStage = [&](int step, int bufI) {
    int r  = step / nCb;
    int cb = (step - r * nCb) << 6;
    int kh = r / ks, kw = r - kh * ks;
    int dy = kh * dil - pad, dx = kw * dil - pad;
    half_t* A = sA + bufI * (128 * CTW2);
    half_t* B = sB + bufI * (64 * CTW2);
    #pragma unroll
    for (int i = 0; i < 4; ++i) {
      int pxl = i * 32 + (tid >> 3);
      int iy = clampi(syv[i] + dy, 0, H - 1);
      int ix = clampi(sxv[i] + dx, 0, W - 1);
      async_cp16(inImg + ((size_t)iy * W + ix) * Cin + cb + qoff, A + pxl * CTW2 + qoff);
    }
    int koff = r * Cin + cb;
    #pragma unroll
    for (int i = 0; i < 2; ++i) {
      int oc = i * 32 + (tid >> 3);
      async_cp16(wpk + (size_t)(ocBlock + oc) * Ktot + koff + qoff, B + oc * CTW2 + qoff);
    }
  };

  issueStage(0, 0);

  for (int st = 0; st < nSteps; ++st) {
    int bufI = st & 1;
    if (st + 1 < nSteps) {
      issueStage(st + 1, bufI ^ 1);
#if HAVE_ASYNC_LDS
      __builtin_amdgcn_s_wait_asynccnt(6);   // stage st complete; st+1 in flight
#endif
    } else {
#if HAVE_ASYNC_LDS
      __builtin_amdgcn_s_wait_asynccnt(0);
#endif
    }
    __syncthreads();

    // fragment validity for this stage's tap
    int r  = st / nCb;
    int kh = r / ks, kw = r - kh * ks;
    int dy = kh * dil - pad, dx = kw * dil - pad;
    int iy0 = ya0 + dy, ix0 = xa0 + dx;
    int iy1 = ya1 + dy, ix1 = xa1 + dx;
    bool v0 = ina0 && iy0 >= 0 && iy0 < H && ix0 >= 0 && ix0 < W;
    bool v1 = ina1 && iy1 >= 0 && iy1 < H && ix1 >= 0 && ix1 < W;

    const half_t* aRow = sA + bufI * (128 * CTW2) + (pxW + lrow) * CTW2;
    const half_t* bRow = sB + bufI * (64 * CTW2) + (ocW + lrow) * CTW2;
    v16h z = zero16h();
    #pragma unroll
    for (int s = 0; s < 2; ++s) {
      int bo = s * 32;
      v16h a0 = frag16(aRow, bo + offA0, bo + offA1);
      v16h a1 = frag16(aRow + 16 * CTW2, bo + offA0, bo + offA1);
      v16h b0 = frag16(bRow, bo + offB0, bo + offB1);
      v16h b1 = frag16(bRow + 16 * CTW2, bo + offB0, bo + offB1);
      a0 = v0 ? a0 : z;
      a1 = v1 ? a1 : z;
      acc00 = __builtin_amdgcn_wmma_f32_16x16x32_f16(false, a0, false, b0, (short)0, acc00, false, false);
      acc01 = __builtin_amdgcn_wmma_f32_16x16x32_f16(false, a0, false, b1, (short)0, acc01, false, false);
      acc10 = __builtin_amdgcn_wmma_f32_16x16x32_f16(false, a1, false, b0, (short)0, acc10, false, false);
      acc11 = __builtin_amdgcn_wmma_f32_16x16x32_f16(false, a1, false, b1, (short)0, acc11, false, false);
    }
    __syncthreads();
  }

  // fused BN(+ReLU) epilogue
  int oc0 = ocBlock + ocW + lrow;
  int oc1 = oc0 + 16;
  float sc0 = gp[oc0] * rsqrtf(vp[oc0] + 1e-5f);
  float sh0 = bp[oc0] - mp[oc0] * sc0;
  float sc1 = gp[oc1] * rsqrtf(vp[oc1] + 1e-5f);
  float sh1 = bp[oc1] - mp[oc1] * sc1;
  half_t* outImg = out + (size_t)n * HW * Cout;
  #pragma unroll
  for (int r = 0; r < 8; ++r) {
    int px0 = pxBlock + pxW + half_id * 8 + r;
    int px1 = px0 + 16;
    if (px0 < HW) {
      float w0 = acc00[r] * sc0 + sh0;
      float w1 = acc01[r] * sc1 + sh1;
      if (relu) { w0 = w0 > 0.f ? w0 : 0.f; w1 = w1 > 0.f ? w1 : 0.f; }
      outImg[(size_t)px0 * Cout + oc0] = (half_t)w0;
      outImg[(size_t)px0 * Cout + oc1] = (half_t)w1;
    }
    if (px1 < HW) {
      float w0 = acc10[r] * sc0 + sh0;
      float w1 = acc11[r] * sc1 + sh1;
      if (relu) { w0 = w0 > 0.f ? w0 : 0.f; w1 = w1 > 0.f ? w1 : 0.f; }
      outImg[(size_t)px1 * Cout + oc0] = (half_t)w0;
      outImg[(size_t)px1 * Cout + oc1] = (half_t)w1;
    }
  }
}

// ---------------- SCNN message passing (sequential per-row GEMM) ----------------
__global__ __launch_bounds__(256)
void mp_wmma_kernel(half_t* __restrict__ buf, const half_t* __restrict__ wT,
                    int H, int W, int seqLen, int lineLen,
                    int strideS, int strideP, int rev) {
  __shared__ __align__(16) half_t sPrev[56 * 128];
  const int tid = threadIdx.x;
  const int lane = tid & 31;
  const int wave = tid >> 5;
  const int n = blockIdx.x;
  const int Ktot = 128 * 9;
  const size_t base = (size_t)n * H * W * 128;

  for (int i = tid; i < 56 * 128; i += 256) sPrev[i] = (half_t)0.0f;
  __syncthreads();
  {
    int s0 = rev ? (seqLen - 1) : 0;
    for (int i = tid; i < lineLen * 128; i += 256) {
      int p = i >> 7, c = i & 127;
      sPrev[(p + 4) * 128 + c] = buf[base + (size_t)s0 * strideS + (size_t)p * strideP + c];
    }
  }
  __syncthreads();

  const int half_id = lane >> 4;
  const int lrow = lane & 15;
  const int numPxTiles = (lineLen + 15) >> 4;
  const int numTiles = numPxTiles * 8;

  for (int step = 1; step < seqLen; ++step) {
    int s = rev ? (seqLen - 1 - step) : step;
    for (int t = wave; t < numTiles; t += 8) {
      int pxT = t % numPxTiles;
      int ocT = t / numPxTiles;
      int pxBase = pxT * 16;
      int ocBase = ocT * 16;
      v8f acc = {0.f, 0.f, 0.f, 0.f, 0.f, 0.f, 0.f, 0.f};
      const half_t* bBase = wT + (size_t)(ocBase + lrow) * Ktot;
      const int m = pxBase + lrow;
      for (int kb = 0; kb < Ktot; kb += 32) {
        int kA0 = kb + half_id * 8;
        int kA1 = kb + 16 + half_id * 8;
        int tap0 = kA0 >> 7, ci0 = kA0 & 127;
        int tap1 = kA1 >> 7, ci1 = kA1 & 127;
        v8h a0 = *(const v8h*)(sPrev + (m + tap0) * 128 + ci0);
        v8h a1 = *(const v8h*)(sPrev + (m + tap1) * 128 + ci1);
        v16h af = __builtin_shufflevector(a0, a1, 0,1,2,3,4,5,6,7,8,9,10,11,12,13,14,15);
        int kB0 = kb + half_id * 8;
        v8h b0 = *(const v8h*)(bBase + kB0);
        v8h b1 = *(const v8h*)(bBase + kB0 + 16);
        v16h bf = __builtin_shufflevector(b0, b1, 0,1,2,3,4,5,6,7,8,9,10,11,12,13,14,15);
        acc = __builtin_amdgcn_wmma_f32_16x16x32_f16(false, af, false, bf, (short)0, acc,
                                                     false, false);
      }
      int oc = ocBase + lrow;
      #pragma unroll
      for (int r = 0; r < 8; ++r) {
        int p = pxBase + half_id * 8 + r;
        if (p < lineLen) {
          size_t gi = base + (size_t)s * strideS + (size_t)p * strideP + oc;
          float cv = acc[r];
          cv = cv > 0.0f ? cv : 0.0f;
          buf[gi] = (half_t)((float)buf[gi] + cv);
        }
      }
    }
    __threadfence_block();
    __syncthreads();
    for (int i = tid; i < lineLen * 128; i += 256) {
      int p = i >> 7, c = i & 127;
      sPrev[(p + 4) * 128 + c] = buf[base + (size_t)s * strideS + (size_t)p * strideP + c];
    }
    __syncthreads();
  }
}

// 1x1 conv 128->1 + bias, f16 NHWC in, f32 out
__global__ void layer2_kernel(const half_t* __restrict__ in, const float* __restrict__ w,
                              const float* __restrict__ b, float* __restrict__ out, int total) {
  int idx = blockIdx.x * blockDim.x + threadIdx.x;
  if (idx >= total) return;
  const half_t* p = in + (size_t)idx * 128;
  float s = 0.0f;
  #pragma unroll 8
  for (int c = 0; c < 128; ++c) s += (float)p[c] * w[c];
  out[idx] = s + b[0];
}

// align-corners bilinear upsample (H,W) -> (outH,outW), f32
__global__ void upsample_kernel(const float* __restrict__ in, float* __restrict__ out,
                                int N, int H, int W, int outH, int outW) {
  size_t idx = (size_t)blockIdx.x * blockDim.x + threadIdx.x;
  size_t total = (size_t)N * outH * outW;
  if (idx >= total) return;
  int ox = idx % outW; size_t r = idx / outW;
  int oy = r % outH; int n = (int)(r / outH);
  float fy = (float)oy * (float)(H - 1) / (float)(outH - 1);
  float fx = (float)ox * (float)(W - 1) / (float)(outW - 1);
  int y0 = (int)floorf(fy); int y1 = y0 + 1 < H ? y0 + 1 : H - 1;
  int x0 = (int)floorf(fx); int x1 = x0 + 1 < W ? x0 + 1 : W - 1;
  float wy = fy - (float)y0, wx = fx - (float)x0;
  const float* img = in + (size_t)n * H * W;
  float v00 = img[y0 * W + x0], v01 = img[y0 * W + x1];
  float v10 = img[y1 * W + x0], v11 = img[y1 * W + x1];
  float top = v00 * (1.0f - wx) + v01 * wx;
  float bot = v10 * (1.0f - wx) + v11 * wx;
  out[idx] = top * (1.0f - wy) + bot * wy;
}

// ---------------- host orchestration ----------------

static inline unsigned cdiv(size_t a, size_t b) { return (unsigned)((a + b - 1) / b); }

extern "C" void kernel_launch(void* const* d_in, const int* in_sizes, int n_in,
                              void* d_out, int out_size, void* d_ws, size_t ws_size,
                              hipStream_t stream) {
  (void)in_sizes; (void)n_in; (void)out_size; (void)ws_size;
  const int NB = 4;
  const float* img = (const float*)d_in[0];

  char* wsBase = (char*)d_ws;
  size_t off = 0;
  auto alloc = [&](size_t bytes) -> void* {
    off = (off + 255) & ~(size_t)255;
    void* p = wsBase + off;
    off += bytes;
    return p;
  };

  const size_t ACT_ELEMS = (size_t)NB * 480 * 640 * 64;
  half_t* actA = (half_t*)alloc(ACT_ELEMS * sizeof(half_t));
  half_t* actB = (half_t*)alloc(ACT_ELEMS * sizeof(half_t));
  float* l2out = (float*)alloc((size_t)NB * 30 * 40 * sizeof(float));

  struct Spec { int cin, cout, H, W, pool, relu; };
  const Spec specs[13] = {
    {3, 64, 480, 640, 0, 1},  {64, 64, 480, 640, 1, 1},
    {64, 128, 240, 320, 0, 1},{128, 128, 240, 320, 1, 1},
    {128, 256, 120, 160, 0, 1},{256, 256, 120, 160, 0, 1},{256, 256, 120, 160, 1, 1},
    {256, 512, 60, 80, 0, 1}, {512, 512, 60, 80, 0, 1},  {512, 512, 60, 80, 1, 1},
    {512, 512, 30, 40, 0, 1}, {512, 512, 30, 40, 0, 1},  {512, 512, 30, 40, 0, 0},
  };

  {
    size_t tot = (size_t)NB * 480 * 640 * 3;
    pack_input_kernel<<<cdiv(tot, 256), 256, 0, stream>>>(img, actA, NB, 480, 640, 3);
  }
  half_t* cur = actA;
  half_t* alt = actB;

  for (int i = 0; i < 13; ++i) {
    const Spec& s = specs[i];
    int K = s.cin * 9;
    half_t* wpk = (half_t*)alloc((size_t)K * s.cout * sizeof(half_t));
    const float* w = (const float*)d_in[1 + 5 * i + 0];
    const float* g = (const float*)d_in[1 + 5 * i + 1];
    const float* b = (const float*)d_in[1 + 5 * i + 2];
    const float* m = (const float*)d_in[1 + 5 * i + 3];
    const float* v = (const float*)d_in[1 + 5 * i + 4];
    pack_convw_kernel<<<cdiv((size_t)K * s.cout, 256), 256, 0, stream>>>(w, wpk, s.cout, s.cin, 3);
    if (i == 0) {
      dim3 grid(cdiv((size_t)s.H * s.W, 64), cdiv(s.cout, 32), NB);
      conv_wmma_kernel<<<grid, 256, 0, stream>>>(cur, wpk, g, b, m, v, alt,
                                                 s.H, s.W, s.cin, s.cout, 3, 1, s.relu);
    } else {
      dim3 grid(cdiv((size_t)s.H * s.W, 128), cdiv(s.cout, 64), NB);
      conv_wmma_fast_kernel<<<grid, 256, 0, stream>>>(cur, wpk, g, b, m, v, alt,
                                                      s.H, s.W, s.cin, s.cout, 3, 1, s.relu);
    }
    half_t* t = cur; cur = alt; alt = t;
    if (s.pool) {
      int Ho = s.H / 2, Wo = s.W / 2;
      size_t tot = (size_t)NB * Ho * Wo * s.cout;
      maxpool_kernel<<<cdiv(tot, 256), 256, 0, stream>>>(cur, alt, NB, s.H, s.W, s.cout);
      t = cur; cur = alt; alt = t;
    }
  }

  // layer1: dilated conv 512->1024 (dil=4) + BN + relu, then 1x1 1024->128 + BN + relu
  {
    const float* w1 = (const float*)d_in[66];
    const float* g1 = (const float*)d_in[67];
    const float* b1 = (const float*)d_in[68];
    const float* m1 = (const float*)d_in[69];
    const float* v1 = (const float*)d_in[70];
    int K1 = 512 * 9;
    half_t* w1pk = (half_t*)alloc((size_t)K1 * 1024 * sizeof(half_t));
    pack_convw_kernel<<<cdiv((size_t)K1 * 1024, 256), 256, 0, stream>>>(w1, w1pk, 1024, 512, 3);
    conv_wmma_fast_kernel<<<dim3(cdiv(1200, 128), 1024 / 64, NB), 256, 0, stream>>>(
        cur, w1pk, g1, b1, m1, v1, alt, 30, 40, 512, 1024, 3, 4, 1);
    half_t* t = cur; cur = alt; alt = t;

    const float* w2 = (const float*)d_in[71];
    const float* g2 = (const float*)d_in[72];
    const float* b2 = (const float*)d_in[73];
    const float* m2 = (const float*)d_in[74];
    const float* v2 = (const float*)d_in[75];
    half_t* w2pk = (half_t*)alloc((size_t)1024 * 128 * sizeof(half_t));
    pack_convw_kernel<<<cdiv((size_t)1024 * 128, 256), 256, 0, stream>>>(w2, w2pk, 128, 1024, 1);
    conv_wmma_fast_kernel<<<dim3(cdiv(1200, 128), 128 / 64, NB), 256, 0, stream>>>(
        cur, w2pk, g2, b2, m2, v2, alt, 30, 40, 1024, 128, 1, 1, 1);
    t = cur; cur = alt; alt = t;
  }

  // message passing: vert-fwd, vert-rev, horiz-fwd, horiz-rev (in place on cur)
  {
    const int dirs[4][5] = {
      {30, 40, 40 * 128, 128, 0},
      {30, 40, 40 * 128, 128, 1},
      {40, 30, 128, 40 * 128, 0},
      {40, 30, 128, 40 * 128, 1},
    };
    for (int d = 0; d < 4; ++d) {
      const float* mw = (const float*)d_in[76 + d];
      half_t* wT = (half_t*)alloc((size_t)128 * 1152 * sizeof(half_t));
      pack_mpw_kernel<<<cdiv(128 * 1152, 256), 256, 0, stream>>>(mw, wT);
      mp_wmma_kernel<<<NB, 256, 0, stream>>>(cur, wT, 30, 40,
                                             dirs[d][0], dirs[d][1], dirs[d][2], dirs[d][3],
                                             dirs[d][4]);
    }
  }

  // layer2 projection + bilinear upsample to d_out
  {
    const float* lw = (const float*)d_in[80];
    const float* lb = (const float*)d_in[81];
    layer2_kernel<<<cdiv(4800, 256), 256, 0, stream>>>(cur, lw, lb, l2out, NB * 30 * 40);
    size_t tot = (size_t)NB * 480 * 640;
    upsample_kernel<<<cdiv(tot, 256), 256, 0, stream>>>(l2out, (float*)d_out, NB, 30, 40, 480, 640);
  }
}
